// ScoreDecoder_62483184222551
// MI455X (gfx1250) — matrix operand order repaired and checked by reading, not compiled, not run on hardware
//
#include <hip/hip_runtime.h>

// ScoreDecoder for MI455X (gfx1250): f16 WMMA GEMMs (f32 accumulate) for all
// 384-wide projections, memory-bound prroi pooling, tiny VALU attention.

typedef _Float16 v8h  __attribute__((ext_vector_type(8)));
typedef _Float16 v16h __attribute__((ext_vector_type(16)));
typedef float    v8f  __attribute__((ext_vector_type(8)));

#define BATCH 512
#define DM    384
#define NH    6
#define HD    64

// ---------------------------------------------------------------- utilities
__device__ inline float block_reduce384(float v, float* red, int tid) {
    red[tid] = v;
    __syncthreads();
    if (tid < 128) red[tid] += red[tid + 128] + red[tid + 256];
    __syncthreads();
    if (tid < 32) {
        float s = red[tid] + red[tid + 32] + red[tid + 64] + red[tid + 96];
        s += __shfl_down(s, 16);
        s += __shfl_down(s, 8);
        s += __shfl_down(s, 4);
        s += __shfl_down(s, 2);
        s += __shfl_down(s, 1);
        if (tid == 0) red[0] = s;
    }
    __syncthreads();
    float r = red[0];
    __syncthreads();
    return r;
}

__device__ inline float Gfun(float t) {
    t = fminf(fmaxf(t, -1.0f), 1.0f);
    return (t < 0.0f) ? 0.5f * (t + 1.0f) * (t + 1.0f)
                      : 1.0f - 0.5f * (1.0f - t) * (1.0f - t);
}

// --------------------------------------------------- PrRoI pooling -> f16 A
// out[(b*16 + py*4+px)*384 + c]  (row-major A for the K/V GEMMs)
__global__ __launch_bounds__(384) void prroi_kernel(
    const float* __restrict__ feat, const float* __restrict__ boxes,
    _Float16* __restrict__ out)
{
    const int b = blockIdx.x, tid = threadIdx.x;
    __shared__ float Wy[4][20], Wx[4][20];
    __shared__ float sInvA;

    const float scale = 20.0f / 320.0f;   // feature-space box
    const float x1 = boxes[b * 4 + 0] * scale;
    const float y1 = boxes[b * 4 + 1] * scale;
    const float x2 = boxes[b * 4 + 2] * scale;
    const float y2 = boxes[b * 4 + 3] * scale;
    const float bw = (x2 - x1) * 0.25f;
    const float bh = (y2 - y1) * 0.25f;

    if (tid < 160) {
        const int which = tid / 80;           // 0 -> Wy, 1 -> Wx
        const int p = (tid % 80) / 20;
        const int i = tid % 20;
        float e0, e1;
        if (which == 0) { e0 = y1 + p * bh; e1 = y1 + (p + 1) * bh; }
        else            { e0 = x1 + p * bw; e1 = x1 + (p + 1) * bw; }
        const float w = Gfun(e1 - (float)i) - Gfun(e0 - (float)i);
        if (which == 0) Wy[p][i] = w; else Wx[p][i] = w;
    }
    if (tid == 0) sInvA = 1.0f / fmaxf(bw * bh, 1e-6f);
    __syncthreads();

    const int c = tid;                       // 384 threads = 384 channels
    const float* f = feat + ((size_t)b * DM + c) * 400;

    float hs[4][20];
#pragma unroll
    for (int p = 0; p < 4; ++p)
#pragma unroll
        for (int w = 0; w < 20; ++w) hs[p][w] = 0.0f;

    for (int h = 0; h < 20; ++h) {
        float row[20];
        const float4* f4 = (const float4*)(f + h * 20);
#pragma unroll
        for (int k = 0; k < 5; ++k) {
            float4 v = f4[k];
            row[4 * k + 0] = v.x; row[4 * k + 1] = v.y;
            row[4 * k + 2] = v.z; row[4 * k + 3] = v.w;
        }
#pragma unroll
        for (int py = 0; py < 4; ++py) {
            const float wy = Wy[py][h];
#pragma unroll
            for (int w = 0; w < 20; ++w) hs[py][w] = fmaf(row[w], wy, hs[py][w]);
        }
    }
    const float inva = sInvA;
#pragma unroll
    for (int py = 0; py < 4; ++py)
#pragma unroll
        for (int px = 0; px < 4; ++px) {
            float acc = 0.0f;
#pragma unroll
            for (int w = 0; w < 20; ++w) acc = fmaf(hs[py][w], Wx[px][w], acc);
            out[((size_t)b * 16 + py * 4 + px) * DM + c] = (_Float16)(acc * inva);
        }
}

// ------------------------------------- template (B,C,64) -> (B*64,384) f16
__global__ __launch_bounds__(256) void tpack_kernel(
    const float* __restrict__ tf, _Float16* __restrict__ out)
{
    const int b = blockIdx.x;
    const float* src = tf + (size_t)b * DM * 64;
    for (int idx = threadIdx.x; idx < DM * 64; idx += 256) {
        const int c = idx >> 6, t = idx & 63;
        out[((size_t)b * 64 + t) * DM + c] = (_Float16)src[idx];
    }
}

// ---------------------- weight (K=384,N=384 row-major f32) -> B-fragment f16
// Wp[((nt*12+kc)*32+lane)*16 + i] = W[K*384+N],
//   K = kc*32 + (lane&16 ? 8 : 0) + (i<8 ? i : 8+i),  N = nt*16 + (lane&15)
__global__ __launch_bounds__(256) void pack_weight_kernel(
    const float* __restrict__ W, _Float16* __restrict__ Wp)
{
    const int idx = blockIdx.x * 256 + threadIdx.x;
    if (idx >= DM * DM) return;
    const int i    = idx & 15;
    const int lane = (idx >> 4) & 31;
    const int kc   = (idx >> 9) % 12;
    const int nt   = idx / (16 * 32 * 12);
    const int K = kc * 32 + ((lane & 16) ? 8 : 0) + (i < 8 ? i : 8 + i);
    const int N = nt * 16 + (lane & 15);
    Wp[idx] = (_Float16)W[(size_t)K * DM + N];
}

// ------------------------------------------------------- WMMA GEMM, N=K=384
// C[M,384] = A[M,384](f16) * W(packed f16) + bias ; optional ReLU.
__global__ __launch_bounds__(256) void gemm384_wmma(
    const _Float16* __restrict__ A, const _Float16* __restrict__ Wp,
    const float* __restrict__ bias, float* __restrict__ outF,
    _Float16* __restrict__ outH, int relu)
{
    __shared__ __align__(16) _Float16 As[16 * DM];     // 12 KB strip of A
    const int row0 = blockIdx.x * 16;
    {
        const unsigned int* Ag  = (const unsigned int*)(A + (size_t)row0 * DM);
        unsigned int*       Asu = (unsigned int*)As;
#pragma unroll
        for (int i = 0; i < 12; ++i)                   // 3072 dwords
            Asu[threadIdx.x + i * 256] = Ag[threadIdx.x + i * 256];
    }
    __syncthreads();

    const int wave = threadIdx.x >> 5;
    const int lane = threadIdx.x & 31;
    const int mrow = lane & 15;
    const int kb   = (lane & 16) ? 8 : 0;              // A-fragment K offset

    v8f acc[3] = {};                                   // 3 N-tiles per wave
    for (int kc = 0; kc < 12; ++kc) {
        const _Float16* ab = As + mrow * DM + kc * 32 + kb;
        v8h alo = *(const v8h*)(ab);                   // K = kc*32+kb .. +7
        v8h ahi = *(const v8h*)(ab + 16);              // K = kc*32+16+kb ..
        v16h a = __builtin_shufflevector(alo, ahi,
                 0, 1, 2, 3, 4, 5, 6, 7, 8, 9, 10, 11, 12, 13, 14, 15);
#pragma unroll
        for (int j = 0; j < 3; ++j) {
            const int nt = wave * 3 + j;
            v16h b = *(const v16h*)(Wp + (((size_t)nt * 12 + kc) * 32 + lane) * 16);
            acc[j] = __builtin_amdgcn_wmma_f32_16x16x32_f16(
                false, a, false, b, (short)0, acc[j], false, false);
        }
    }

    const int mb = row0 + ((lane & 16) ? 8 : 0);
#pragma unroll
    for (int j = 0; j < 3; ++j) {
        const int nt   = wave * 3 + j;
        const int ncol = nt * 16 + (lane & 15);
        const float bv = bias ? bias[ncol] : 0.0f;
#pragma unroll
        for (int r = 0; r < 8; ++r) {
            float v = acc[j][r] + bv;
            if (relu) v = fmaxf(v, 0.0f);
            const size_t off = (size_t)(mb + r) * DM + ncol;
            if (outF) outF[off] = v;
            if (outH) outH[off] = (_Float16)v;
        }
    }
}

// ---------------------------- score token: layernorm + q0 = x0 @ Wq0 + bq0
__global__ __launch_bounds__(384) void q0_kernel(
    const float* __restrict__ tok, const float* __restrict__ g,
    const float* __restrict__ bt, const float* __restrict__ Wq,
    const float* __restrict__ bq, float* __restrict__ q0)
{
    __shared__ float red[384];
    __shared__ float x0[384];
    const int tid = threadIdx.x;
    const float v = tok[tid];
    const float m = block_reduce384(v, red, tid) * (1.0f / 384.0f);
    const float d = v - m;
    const float var = block_reduce384(d * d, red, tid) * (1.0f / 384.0f);
    x0[tid] = d * rsqrtf(var + 1e-5f) * g[tid] + bt[tid];
    __syncthreads();
    float acc = bq[tid];
    for (int c = 0; c < DM; ++c) acc = fmaf(x0[c], Wq[(size_t)c * DM + tid], acc);
    q0[tid] = acc;
}

// ----------------------------- single-query attention (per batch), T keys
__global__ __launch_bounds__(384) void attn_kernel(
    const _Float16* __restrict__ K, const _Float16* __restrict__ V,
    const float* __restrict__ q, int qstride, int T,
    _Float16* __restrict__ outA)
{
    const int b = blockIdx.x, tid = threadIdx.x;
    __shared__ float lg[NH * 64];
    const float scaleAttn = 0.05103103630798288f;      // 384^-0.5

    const int h = tid % NH, t = tid / NH;
    if (t < T) {
        const float*    qp = q + (size_t)b * qstride + h * HD;
        const _Float16* kp = K + ((size_t)b * T + t) * DM + h * HD;
        float acc = 0.0f;
#pragma unroll 8
        for (int d = 0; d < HD; ++d) acc = fmaf(qp[d], (float)kp[d], acc);
        lg[h * 64 + t] = acc * scaleAttn;
    }
    __syncthreads();
    if (tid < NH) {
        float m = -1e30f;
        for (int i = 0; i < T; ++i) m = fmaxf(m, lg[tid * 64 + i]);
        float s = 0.0f;
        for (int i = 0; i < T; ++i) {
            const float e = __expf(lg[tid * 64 + i] - m);
            lg[tid * 64 + i] = e;
            s += e;
        }
        const float inv = 1.0f / s;
        for (int i = 0; i < T; ++i) lg[tid * 64 + i] *= inv;
    }
    __syncthreads();
    const int c = tid, hc = c >> 6;
    float acc = 0.0f;
    for (int t2 = 0; t2 < T; ++t2)
        acc = fmaf(lg[hc * 64 + t2], (float)V[((size_t)b * T + t2) * DM + c], acc);
    outA[(size_t)b * DM + c] = (_Float16)acc;
}

// ------------------------------------------------ layernorm (B,384) -> f16
__global__ __launch_bounds__(384) void ln_kernel(
    const float* __restrict__ X, const float* __restrict__ g,
    const float* __restrict__ bt, _Float16* __restrict__ outH)
{
    __shared__ float red[384];
    const int b = blockIdx.x, tid = threadIdx.x;
    const float v = X[(size_t)b * DM + tid];
    const float m = block_reduce384(v, red, tid) * (1.0f / 384.0f);
    const float d = v - m;
    const float var = block_reduce384(d * d, red, tid) * (1.0f / 384.0f);
    outH[(size_t)b * DM + tid] =
        (_Float16)(d * rsqrtf(var + 1e-5f) * g[tid] + bt[tid]);
}

// ------------------------------------------------------ final 384 -> 1 dot
__global__ __launch_bounds__(384) void final_kernel(
    const _Float16* __restrict__ h1, const float* __restrict__ W2,
    const float* __restrict__ b2, float* __restrict__ out)
{
    __shared__ float red[384];
    const int b = blockIdx.x, tid = threadIdx.x;
    const float v = (float)h1[(size_t)b * DM + tid] * W2[tid];
    const float s = block_reduce384(v, red, tid);
    if (tid == 0) out[b] = s + b2[0];
}

// ============================================================== launcher
extern "C" void kernel_launch(void* const* d_in, const int* in_sizes, int n_in,
                              void* d_out, int out_size, void* d_ws, size_t ws_size,
                              hipStream_t stream) {
    (void)in_sizes; (void)n_in; (void)out_size; (void)ws_size;

    const float* search_feat   = (const float*)d_in[0];
    const float* template_feat = (const float*)d_in[1];
    const float* search_box    = (const float*)d_in[2];
    const float* score_token   = (const float*)d_in[3];
    const float* norm1_g       = (const float*)d_in[4];
    const float* norm1_b       = (const float*)d_in[5];
    const float* mlp_W0        = (const float*)d_in[6];
    const float* mlp_b0        = (const float*)d_in[7];
    const float* mlp_W1        = (const float*)d_in[8];
    const float* mlp_b1        = (const float*)d_in[9];
    const float* mlp_W2        = (const float*)d_in[10];
    const float* mlp_b2        = (const float*)d_in[11];
    // stage dicts: Wq,bq,Wk,bk,Wv,bv,Wo,bo,norm_g,norm_b
    const float* Wq0 = (const float*)d_in[12]; const float* bq0 = (const float*)d_in[13];
    const float* Wk0 = (const float*)d_in[14]; const float* bk0 = (const float*)d_in[15];
    const float* Wv0 = (const float*)d_in[16]; const float* bv0 = (const float*)d_in[17];
    const float* Wo0 = (const float*)d_in[18]; const float* bo0 = (const float*)d_in[19];
    const float* ng0 = (const float*)d_in[20]; const float* nb0 = (const float*)d_in[21];
    const float* Wq1 = (const float*)d_in[22]; const float* bq1 = (const float*)d_in[23];
    const float* Wk1 = (const float*)d_in[24]; const float* bk1 = (const float*)d_in[25];
    const float* Wv1 = (const float*)d_in[26]; const float* bv1 = (const float*)d_in[27];
    const float* Wo1 = (const float*)d_in[28]; const float* bo1 = (const float*)d_in[29];
    const float* ng1 = (const float*)d_in[30]; const float* nb1 = (const float*)d_in[31];

    float* out = (float*)d_out;

    // workspace carve-up (256-B aligned)
    char* p = (char*)d_ws;
    auto alloc = [&](size_t bytes) -> void* {
        void* r = (void*)p;
        p += (bytes + 255) & ~(size_t)255;
        return r;
    };
    const size_t SBF = (size_t)BATCH * 16 * DM;   // 8192 x 384
    const size_t TFE = (size_t)BATCH * 64 * DM;   // 32768 x 384
    const size_t BD  = (size_t)BATCH * DM;        // 512 x 384
    const size_t WPN = (size_t)DM * DM;           // packed weight elems

    _Float16* sbf_h  = (_Float16*)alloc(SBF * 2);
    _Float16* tf_h   = (_Float16*)alloc(TFE * 2);
    _Float16* wp     = (_Float16*)alloc(9 * WPN * 2);
    float*    q0     = (float*)   alloc(DM * 4);
    _Float16* K0h    = (_Float16*)alloc(SBF * 2);
    _Float16* V0h    = (_Float16*)alloc(SBF * 2);
    _Float16* K1h    = (_Float16*)alloc(TFE * 2);
    _Float16* V1h    = (_Float16*)alloc(TFE * 2);
    _Float16* attn_h = (_Float16*)alloc(BD * 2);
    float*    wo_f   = (float*)   alloc(BD * 4);
    float*    q1_f   = (float*)   alloc(BD * 4);
    _Float16* x1_h   = (_Float16*)alloc(BD * 2);
    _Float16* x2_h   = (_Float16*)alloc(BD * 2);
    _Float16* h0_h   = (_Float16*)alloc(BD * 2);
    _Float16* h1_h   = (_Float16*)alloc(BD * 2);

    _Float16* wpWk0 = wp + 0 * WPN; _Float16* wpWv0 = wp + 1 * WPN;
    _Float16* wpWo0 = wp + 2 * WPN; _Float16* wpWq1 = wp + 3 * WPN;
    _Float16* wpWk1 = wp + 4 * WPN; _Float16* wpWv1 = wp + 5 * WPN;
    _Float16* wpWo1 = wp + 6 * WPN; _Float16* wpW0  = wp + 7 * WPN;
    _Float16* wpW1  = wp + 8 * WPN;

    // stage A: inputs -> f16 GEMM operands
    prroi_kernel<<<BATCH, 384, 0, stream>>>(search_feat, search_box, sbf_h);
    tpack_kernel<<<BATCH, 256, 0, stream>>>(template_feat, tf_h);
    const float* packSrc[9] = {Wk0, Wv0, Wo0, Wq1, Wk1, Wv1, Wo1, mlp_W0, mlp_W1};
    for (int i = 0; i < 9; ++i)
        pack_weight_kernel<<<(DM * DM + 255) / 256, 256, 0, stream>>>(
            packSrc[i], wp + (size_t)i * WPN);
    q0_kernel<<<1, 384, 0, stream>>>(score_token, norm1_g, norm1_b, Wq0, bq0, q0);

    // stage 0: K/V over pooled search region (16 tokens)
    gemm384_wmma<<<SBF / (16 * DM) * 1, 256, 0, stream>>>(sbf_h, wpWk0, bk0, nullptr, K0h, 0);
    gemm384_wmma<<<SBF / (16 * DM) * 1, 256, 0, stream>>>(sbf_h, wpWv0, bv0, nullptr, V0h, 0);
    attn_kernel<<<BATCH, 384, 0, stream>>>(K0h, V0h, q0, 0, 16, attn_h);
    gemm384_wmma<<<BATCH / 16, 256, 0, stream>>>(attn_h, wpWo0, bo0, wo_f, nullptr, 0);
    ln_kernel<<<BATCH, 384, 0, stream>>>(wo_f, ng0, nb0, x1_h);

    // stage 1: K/V over template tokens (64 tokens)
    gemm384_wmma<<<BATCH / 16, 256, 0, stream>>>(x1_h, wpWq1, bq1, q1_f, nullptr, 0);
    gemm384_wmma<<<TFE / (16 * DM) * 1, 256, 0, stream>>>(tf_h, wpWk1, bk1, nullptr, K1h, 0);
    gemm384_wmma<<<TFE / (16 * DM) * 1, 256, 0, stream>>>(tf_h, wpWv1, bv1, nullptr, V1h, 0);
    attn_kernel<<<BATCH, 384, 0, stream>>>(K1h, V1h, q1_f, DM, 64, attn_h);
    gemm384_wmma<<<BATCH / 16, 256, 0, stream>>>(attn_h, wpWo1, bo1, wo_f, nullptr, 0);
    ln_kernel<<<BATCH, 384, 0, stream>>>(wo_f, ng1, nb1, x2_h);

    // MLP head
    gemm384_wmma<<<BATCH / 16, 256, 0, stream>>>(x2_h, wpW0, mlp_b0, nullptr, h0_h, 1);
    gemm384_wmma<<<BATCH / 16, 256, 0, stream>>>(h0_h, wpW1, mlp_b1, nullptr, h1_h, 1);
    final_kernel<<<BATCH, 384, 0, stream>>>(h1_h, mlp_W2, mlp_b2, out);
}
